// DalleSelfAttention_60318520705234
// MI455X (gfx1250) — matrix operand order repaired and checked by reading, not compiled, not run on hardware
//
#include <hip/hip_runtime.h>
#include <cmath>

// ---------------------------------------------------------------------------
// DALLE-style self attention for MI455X (gfx1250, wave32, WMMA).
// QKV proj, attention (flash), output proj all on v_wmma_f32_16x16x32_bf16
// with f32 accumulation. All LDS fragment reads are ds_load_b128 (B-operands
// stored K-contiguous in LDS, 16B-multiple strides). Global->LDS staging is
// software-pipelined through registers with ping-pong LDS buffers so global
// latency overlaps the WMMAs; one barrier per GEMM K-step.
// Workspace: qkv [B,S,3H] f32 + ctx [B,S,H] f32  (~67 MB).
// ---------------------------------------------------------------------------

typedef __bf16 bf16_t;
typedef bf16_t v4bf  __attribute__((ext_vector_type(4)));
typedef bf16_t v8bf  __attribute__((ext_vector_type(8)));
typedef bf16_t v16bf __attribute__((ext_vector_type(16)));
typedef float  v8f   __attribute__((ext_vector_type(8)));

constexpr int B_  = 2;
constexpr int S_  = 2048;
constexpr int H_  = 1024;
constexpr int N_  = 16;
constexpr int HN_ = 64;

__device__ __forceinline__ v16bf cat8(v8bf lo, v8bf hi) {
  return __builtin_shufflevector(lo, hi, 0, 1, 2, 3, 4, 5, 6, 7,
                                 8, 9, 10, 11, 12, 13, 14, 15);
}

// A fragment 16x32 (MxK), row-major bf16 source, ld = multiple of 8.
// lanes 0-15: row=lane, K={0..7,16..23}; lanes 16-31: row=lane-16, K={8..15,24..31}
__device__ __forceinline__ v16bf frag_a(const bf16_t* src, int ld, int lane) {
  const bf16_t* p = src + (lane & 15) * ld + ((lane >> 4) << 3);
  const v8bf lo = *(const v8bf*)(p);
  const v8bf hi = *(const v8bf*)(p + 16);
  return cat8(lo, hi);
}

// B fragment 32x16 (KxN) from K-contiguous storage src[n][k], ld = multiple of 8.
// lanes 0-15: col=lane, K={0..7,16..23}; lanes 16-31: col=lane-16, K={8..15,24..31}
__device__ __forceinline__ v16bf frag_bk(const bf16_t* src, int ld, int lane) {
  const bf16_t* p = src + (lane & 15) * ld + ((lane >> 4) << 3);
  const v8bf lo = *(const v8bf*)(p);
  const v8bf hi = *(const v8bf*)(p + 16);
  return cat8(lo, hi);
}

__device__ __forceinline__ v8f wmma_bf16(v16bf a, v16bf b, v8f c) {
  return __builtin_amdgcn_wmma_f32_16x16x32_bf16(false, a, false, b, (short)0, c,
                                                 false, false);
}

__device__ __forceinline__ v4bf cvt4(float4 f) {
  v4bf r;
  r[0] = (bf16_t)f.x; r[1] = (bf16_t)f.y; r[2] = (bf16_t)f.z; r[3] = (bf16_t)f.w;
  return r;
}

// ---- GEMM + bias:  C[M,N] = A[M,K] @ W[K,N] + bias[N]  (f32 io) ------------
// Block tile 64x128, 8 waves (wm 0..3 x wn 0..1), each wave 16x64 out.
// K-step 32; register-staged double-buffered LDS; 1 barrier per K-step.
__global__ __launch_bounds__(256) void gemm_bias_kernel(
    const float* __restrict__ A, const float* __restrict__ W,
    const float* __restrict__ bias, float* __restrict__ C,
    int M, int N, int K) {
  constexpr int TM = 64, TN = 128, TK = 32;
  constexpr int LDA = TK + 8;  // 40 bf16 = 80B rows (16B multiple)
  constexpr int LDB = TK + 8;  // Bs stored transposed: [n][k]
  __shared__ __align__(16) bf16_t As[2][TM * LDA];
  __shared__ __align__(16) bf16_t Bs[2][TN * LDB];

  const int block_m = blockIdx.y * TM;
  const int block_n = blockIdx.x * TN;
  const int tid  = threadIdx.x;
  const int lane = tid & 31;
  const int wave = tid >> 5;   // 0..7
  const int wm   = wave & 3;   // 16-row slice
  const int wn   = wave >> 2;  // 64-col slice

  // staging coordinates
  const int ar0 = (tid) >> 3,        ac0 = (tid & 7) * 4;        // A vec4 #tid
  const int ar1 = (256 + tid) >> 3,  ac1 = ((256 + tid) & 7) * 4;
  const int nn = tid & 127;
  const int kq = tid >> 7;     // 0..1 -> k offset 0 / 16

  float4 ra0, ra1;
  float  rb[16];

  auto load_tile = [&](int k0) {
    ra0 = *(const float4*)&A[(size_t)(block_m + ar0) * K + (k0 + ac0)];
    ra1 = *(const float4*)&A[(size_t)(block_m + ar1) * K + (k0 + ac1)];
    const float* wp = &W[(size_t)(k0 + kq * 16) * N + (block_n + nn)];
#pragma unroll
    for (int kk = 0; kk < 16; ++kk) rb[kk] = wp[(size_t)kk * N];
  };
  auto store_tile = [&](int buf) {
    *(v4bf*)&As[buf][ar0 * LDA + ac0] = cvt4(ra0);
    *(v4bf*)&As[buf][ar1 * LDA + ac1] = cvt4(ra1);
#pragma unroll
    for (int g = 0; g < 4; ++g) {
      v4bf t;
#pragma unroll
      for (int kk = 0; kk < 4; ++kk) t[kk] = (bf16_t)rb[g * 4 + kk];
      *(v4bf*)&Bs[buf][nn * LDB + kq * 16 + g * 4] = t;
    }
  };

  v8f acc[4] = {};

  load_tile(0);
  store_tile(0);
  const int nsteps = K / TK;
  for (int s = 0; s < nsteps; ++s) {
    __syncthreads();                         // staged buffer (s&1) ready
    if (s + 1 < nsteps) load_tile((s + 1) * TK);  // overlap with WMMAs below
    const int cur = s & 1;
    const v16bf a = frag_a(&As[cur][(wm * 16) * LDA], LDA, lane);
#pragma unroll
    for (int t = 0; t < 4; ++t) {
      const v16bf b = frag_bk(&Bs[cur][(wn * 64 + t * 16) * LDB], LDB, lane);
      acc[t] = wmma_bf16(a, b, acc[t]);
    }
    if (s + 1 < nsteps) store_tile(cur ^ 1);
  }

  // epilogue: C-layout row = e + 8*(lane>=16), col = lane&15
  const int hi = (lane >> 4) ? 8 : 0;
  const int cl = lane & 15;
#pragma unroll
  for (int t = 0; t < 4; ++t) {
    const int col = block_n + wn * 64 + t * 16 + cl;
    const float bv = bias[col];
#pragma unroll
    for (int e = 0; e < 8; ++e) {
      const int row = block_m + wm * 16 + e + hi;
      C[(size_t)row * N + col] = acc[t][e] + bv;
    }
  }
}

// ---- Flash attention over qkv buffer --------------------------------------
// grid: (S/64 q-blocks, B*N heads), block: 128 (4 waves, 16 q-rows each).
__global__ __launch_bounds__(128) void attn_kernel(const float* __restrict__ qkv,
                                                   float* __restrict__ ctx) {
  constexpr int LDQ = HN_ + 8;  // 72 bf16 = 144B rows (16B multiple)
  constexpr int LDV = 32 + 8;   // Vs rows: 40 bf16
  constexpr int LDP = 32 + 8;
  __shared__ __align__(16) bf16_t Qs[64 * LDQ];      // [q][d] row-major
  __shared__ __align__(16) bf16_t Ks[2][32 * LDQ];   // [key][d] row-major (=B^T)
  __shared__ __align__(16) bf16_t Vs[2][HN_ * LDV];  // transposed: [d][key]
  __shared__ __align__(16) bf16_t Ps[4][16 * LDP];

  const int qb   = blockIdx.x;
  const int bh   = blockIdx.y;
  const int b    = bh / N_;
  const int n    = bh % N_;
  const int tid  = threadIdx.x;
  const int lane = tid & 31;
  const int wave = tid >> 5;

  const size_t rs = 3 * H_;  // qkv row stride in elems
  const float* qptr = qkv + (size_t)(b * S_) * rs + n * HN_;
  const float* kptr = qptr + H_;
  const float* vptr = qptr + 2 * H_;

  // stage Q (64 x 64): float4 loads, v4bf stores
#pragma unroll
  for (int i = 0; i < 8; ++i) {
    const int idx = i * 128 + tid;           // 0..1023 float4's
    const int r = idx >> 4, c4 = idx & 15;
    const float4 f = *(const float4*)&qptr[(size_t)(qb * 64 + r) * rs + c4 * 4];
    *(v4bf*)&Qs[r * LDQ + c4 * 4] = cvt4(f);
  }
  __syncthreads();

  const v16bf qa0 = frag_a(&Qs[(wave * 16) * LDQ + 0],  LDQ, lane);
  const v16bf qa1 = frag_a(&Qs[(wave * 16) * LDQ + 32], LDQ, lane);

  v8f acc0 = {}, acc1 = {}, acc2 = {}, acc3 = {};
  float m_i[8], l_i[8];
#pragma unroll
  for (int e = 0; e < 8; ++e) { m_i[e] = -1e30f; l_i[e] = 0.0f; }

  const int hi  = (lane >> 4) ? 8 : 0;
  const int col = lane & 15;
  // K staging: 4 float4 per thread; V staging: thread owns dim d, 16 keys
  const int vd = tid & 63;
  const int vk = tid >> 6;  // 0..1 -> key offset 0 / 16

  float4 rk[4];
  float  rv[16];

  auto load_kv = [&](int kb) {
#pragma unroll
    for (int i = 0; i < 4; ++i) {
      const int idx = i * 128 + tid;         // 0..511 float4's of K tile
      const int r = idx >> 4, c4 = idx & 15;
      rk[i] = *(const float4*)&kptr[(size_t)(kb + r) * rs + c4 * 4];
    }
    const float* vp = &vptr[(size_t)(kb + vk * 16) * rs + vd];
#pragma unroll
    for (int kk = 0; kk < 16; ++kk) rv[kk] = vp[(size_t)kk * rs];
  };
  auto store_kv = [&](int buf) {
#pragma unroll
    for (int i = 0; i < 4; ++i) {
      const int idx = i * 128 + tid;
      const int r = idx >> 4, c4 = idx & 15;
      *(v4bf*)&Ks[buf][r * LDQ + c4 * 4] = cvt4(rk[i]);
    }
#pragma unroll
    for (int g = 0; g < 4; ++g) {
      v4bf t;
#pragma unroll
      for (int kk = 0; kk < 4; ++kk) t[kk] = (bf16_t)rv[g * 4 + kk];
      *(v4bf*)&Vs[buf][vd * LDV + vk * 16 + g * 4] = t;
    }
  };

  const int nkb = 2 * qb + 2;  // causal: key blocks of 32 up to qb*64+63

  load_kv(0);
  store_kv(0);
  for (int j = 0; j < nkb; ++j) {
    __syncthreads();                        // staged buffer (j&1) ready
    const int kb = j * 32;
    const int cur = j & 1;
    if (j + 1 < nkb) load_kv(kb + 32);      // overlap with compute below

    // scores S = Q . K^T : two 16x16 key tiles, K-dim 64 -> 2 WMMAs each
    v8f s0 = {}, s1 = {};
    s0 = wmma_bf16(qa0, frag_bk(&Ks[cur][0 * LDQ + 0],   LDQ, lane), s0);
    s0 = wmma_bf16(qa1, frag_bk(&Ks[cur][0 * LDQ + 32],  LDQ, lane), s0);
    s1 = wmma_bf16(qa0, frag_bk(&Ks[cur][16 * LDQ + 0],  LDQ, lane), s1);
    s1 = wmma_bf16(qa1, frag_bk(&Ks[cur][16 * LDQ + 32], LDQ, lane), s1);

    // online softmax; row r of a C tile lives in one 16-lane half
#pragma unroll
    for (int e = 0; e < 8; ++e) {
      const int qrow = qb * 64 + wave * 16 + e + hi;
      const float v0 = ((kb + col)      <= qrow) ? s0[e] * 0.125f : -1.0e4f;
      const float v1 = ((kb + 16 + col) <= qrow) ? s1[e] * 0.125f : -1.0e4f;
      float smax = fmaxf(v0, v1);
#pragma unroll
      for (int off = 1; off < 16; off <<= 1)
        smax = fmaxf(smax, __shfl_xor(smax, off, 32));
      const float mn = fmaxf(m_i[e], smax);
      const float sc = __expf(m_i[e] - mn);
      const float p0 = __expf(v0 - mn);
      const float p1 = __expf(v1 - mn);
      float rsum = p0 + p1;
#pragma unroll
      for (int off = 1; off < 16; off <<= 1)
        rsum += __shfl_xor(rsum, off, 32);
      l_i[e] = l_i[e] * sc + rsum;
      m_i[e] = mn;
      acc0[e] *= sc; acc1[e] *= sc; acc2[e] *= sc; acc3[e] *= sc;
      // C-layout -> A-layout transpose goes through LDS
      const int m = e + hi;
      Ps[wave][m * LDP + col]      = (bf16_t)p0;
      Ps[wave][m * LDP + 16 + col] = (bf16_t)p1;
    }
    __syncthreads();                        // Ps ready

    // ctx += P(16x32) . V(32x64); Vs[d][key] is K-contiguous for frag_bk
    const v16bf pa = frag_a(&Ps[wave][0], LDP, lane);
    acc0 = wmma_bf16(pa, frag_bk(&Vs[cur][0  * LDV], LDV, lane), acc0);
    acc1 = wmma_bf16(pa, frag_bk(&Vs[cur][16 * LDV], LDV, lane), acc1);
    acc2 = wmma_bf16(pa, frag_bk(&Vs[cur][32 * LDV], LDV, lane), acc2);
    acc3 = wmma_bf16(pa, frag_bk(&Vs[cur][48 * LDV], LDV, lane), acc3);

    if (j + 1 < nkb) store_kv(cur ^ 1);     // fill next buffer
  }

  // normalize and store ctx[b, q, n*64 + d] (f32)
#pragma unroll
  for (int e = 0; e < 8; ++e) {
    const int qrow = qb * 64 + wave * 16 + e + hi;
    const float inv = 1.0f / l_i[e];
    float* dst = ctx + (size_t)(b * S_ + qrow) * H_ + n * HN_;
    dst[col]      = acc0[e] * inv;
    dst[16 + col] = acc1[e] * inv;
    dst[32 + col] = acc2[e] * inv;
    dst[48 + col] = acc3[e] * inv;
  }
}

// ---------------------------------------------------------------------------

extern "C" void kernel_launch(void* const* d_in, const int* in_sizes, int n_in,
                              void* d_out, int out_size, void* d_ws, size_t ws_size,
                              hipStream_t stream) {
  const float* hidden  = (const float*)d_in[0];
  // d_in[1] = ltor_mask : causal mask computed inline, input unused.
  const float* w_qkv   = (const float*)d_in[2];
  const float* b_qkv   = (const float*)d_in[3];
  const float* w_dense = (const float*)d_in[4];
  const float* b_dense = (const float*)d_in[5];
  float* out = (float*)d_out;

  float* qkv = (float*)d_ws;                    // [B,S,3H]  50.3 MB
  float* ctx = qkv + (size_t)B_ * S_ * 3 * H_;  // [B,S,H]   16.8 MB

  const int M = B_ * S_;  // 4096

  // 1) QKV projection: [4096,1024] x [1024,3072] + bias
  gemm_bias_kernel<<<dim3((3 * H_) / 128, M / 64), 256, 0, stream>>>(
      hidden, w_qkv, b_qkv, qkv, M, 3 * H_, H_);

  // 2) causal flash attention per (head, 64-row q block)
  attn_kernel<<<dim3(S_ / 64, B_ * N_), 128, 0, stream>>>(qkv, ctx);

  // 3) output projection: [4096,1024] x [1024,1024] + bias
  gemm_bias_kernel<<<dim3(H_ / 128, M / 64), 256, 0, stream>>>(
      ctx, w_dense, b_dense, out, M, H_, H_);
}
